// BasicBlock3_4947802325426
// MI455X (gfx1250) — matrix operand order repaired and checked by reading, not compiled, not run on hardware
//
#include <hip/hip_runtime.h>
#include <hip/hip_bf16.h>
#include <stdint.h>

// ---------------------------------------------------------------------------
// Deformable-conv BasicBlock for MI455X (gfx1250, wave32, WMMA).
//
// All six stages are GEMMs over N = B*H*W = 25088 positions, C = 64 channels:
// bilinear-gathered im2col columns (f16) x pre-packed f16 weights via
// V_WMMA_F32_16X16X32_F16 with f32 accumulation. Total footprint ~16 MB
// (fits the 192 MB L2); the bottleneck is gather VALU/VMEM + matrix pipe,
// so we double-buffer the LDS column chunk to overlap gather(t+1) with
// WMMA(t), and use b128 LDS traffic for the fragments.
// ---------------------------------------------------------------------------

typedef __attribute__((ext_vector_type(16))) _Float16 v16h;
typedef __attribute__((ext_vector_type(8)))  float    v8f;
typedef __attribute__((ext_vector_type(4)))  uint32_t u32x4;

#define HW  3136   // 56*56
#define WID 56
#define CIN 64
#define LDSROW 36  // padded row stride (dwords): 16B-aligned rows, bank-spread

union FragU { uint32_t u[8]; u32x4 q[2]; v16h h; };
union PackU { _Float16 h[2]; uint32_t u; };

__device__ __forceinline__ float relu_f(float x) { return x > 0.f ? x : 0.f; }

// ---------------------------------------------------------------------------
// Weight packing: src (O, 64, KH, KW) f32 -> dst[K2][32][64] u32 (half2).
// dst[tap][cpair][m] = half2( w[m][2*cpair][tap], w[m][2*cpair+1][tap] ),
// rows m >= Cout zero-padded. This is the direct A-fragment feed layout.
// ---------------------------------------------------------------------------
__global__ void pack_w_kernel(const float* __restrict__ src, uint32_t* __restrict__ dst,
                              int Cout, int K2) {
    int idx = blockIdx.x * blockDim.x + threadIdx.x;
    int total = K2 * 32 * 64;
    if (idx >= total) return;
    int tap   = idx >> 11;
    int r     = idx & 2047;
    int cpair = r >> 6;
    int m     = r & 63;
    float v0 = 0.f, v1 = 0.f;
    if (m < Cout) {
        int c0 = cpair * 2;
        v0 = src[(size_t)(m * CIN + c0) * K2 + tap];
        v1 = src[(size_t)(m * CIN + c0 + 1) * K2 + tap];
    }
    PackU p; p.h[0] = (_Float16)v0; p.h[1] = (_Float16)v1;
    dst[idx] = p.u;
}

// Fold BN: scale = g / sqrt(v + eps), shift = b - mean * scale.
__global__ void bn_prep_kernel(const float* __restrict__ g, const float* __restrict__ b,
                               const float* __restrict__ m, const float* __restrict__ v,
                               float* __restrict__ scale, float* __restrict__ shift) {
    int i = threadIdx.x;
    if (i < 64) {
        float inv = g[i] * __frsqrt_rn(v[i] + 1e-5f);
        scale[i] = inv;
        shift[i] = b[i] - m[i] * inv;
    }
}

// ---------------------------------------------------------------------------
// Unified (deformable) conv kernel.
//   mode : 0 = +bias | 1 = relu(bn(.)) | 2 = out += relu(bn(.))
//          3 = relu(bn(.) + residual)
// Block: 128 threads = 4 waves. Tile: 64 positions x 64 (padded) out-channels.
// LDS: double-buffered 64(n) x 32(cpair) f16-pair chunks, row stride 36 dwords.
// Pipeline per tap: gather(t+1) -> wmma(t) -> barrier (1 barrier/tap).
// ---------------------------------------------------------------------------
template <int KH, int KW, bool DEFORM>
__global__ __launch_bounds__(128)
void deform_conv_wmma(const float* __restrict__ in,
                      const float* __restrict__ offs,
                      const uint32_t* __restrict__ wpk,
                      float* __restrict__ out,
                      int Mout, int pad, int dil, int mode,
                      const float* __restrict__ bias,
                      const float* __restrict__ scale,
                      const float* __restrict__ shift,
                      const float* __restrict__ residual) {
    constexpr int K2 = KH * KW;

    __shared__ __align__(16) uint32_t ldsB[2][64 * LDSROW];  // 2 x 9 KB

    const int t      = threadIdx.x;
    const int lane   = t & 31;
    const int waveId = t >> 5;
    const int mtile  = waveId;                 // 16 output rows per wave
    const bool waveActive = (mtile * 16) < Mout;

    const int nBase = blockIdx.x * 64;         // 3136 % 64 == 0 -> one batch img
    const int b     = nBase / HW;
    const float* __restrict__ xb = in + (size_t)b * CIN * HW;

    // Gather role: thread owns one position gn (0..63) and 32 channels (half).
    const int gn    = t & 63;
    const int ghalf = t >> 6;                  // 0/1: channels 0-31 / 32-63
    const int rem_g = nBase + gn - b * HW;
    const int h_g   = rem_g / WID;
    const int w_g   = rem_g % WID;

    const int ln15 = lane & 15;
    const int hi   = lane >> 4;

    v8f acc[4];
    #pragma unroll
    for (int nt = 0; nt < 4; ++nt) acc[nt] = (v8f){0.f,0.f,0.f,0.f,0.f,0.f,0.f,0.f};

    // ---- cooperative bilinear gather of one tap's 64(c) x 64(n) chunk ----
    auto gather = [&](int tap, uint32_t* __restrict__ bufp) {
        const int ki = tap / KW;
        const int kj = tap % KW;
        float dy = 0.f, dx = 0.f;
        if (DEFORM) {
            const float* op = offs + ((size_t)(b * 2 * K2 + 2 * tap)) * HW + rem_g;
            dy = op[0];
            dx = op[HW];
            if (tap + 1 < K2) __builtin_prefetch(op + 2 * HW, 0, 0);  // next tap's dy
        }
        const float py = dy + (float)(ki * dil - pad) + (float)h_g;
        const float px = dx + (float)(kj * dil - pad) + (float)w_g;
        const float y0f = floorf(py), x0f = floorf(px);
        const int   y0  = (int)y0f,   x0  = (int)x0f;
        const float fy  = py - y0f,   fx  = px - x0f;

        float cw[4];
        int   cidx[4];
        #pragma unroll
        for (int cn = 0; cn < 4; ++cn) {
            const int yc = y0 + (cn >> 1);
            const int xc = x0 + (cn & 1);
            const bool valid = (yc >= 0) & (yc < WID) & (xc >= 0) & (xc < WID);
            const float wy = (cn >> 1) ? fy : (1.f - fy);
            const float wx = (cn & 1)  ? fx : (1.f - fx);
            cw[cn]   = valid ? wy * wx : 0.f;
            const int yi = yc < 0 ? 0 : (yc > WID - 1 ? WID - 1 : yc);
            const int xi = xc < 0 ? 0 : (xc > WID - 1 ? WID - 1 : xc);
            cidx[cn] = yi * WID + xi;
        }

        uint32_t pk[16];
        #pragma unroll
        for (int i = 0; i < 16; ++i) {
            const float* p0 = xb + (size_t)(ghalf * 32 + 2 * i) * HW;
            const float* p1 = p0 + HW;
            float v0 = cw[0] * p0[cidx[0]] + cw[1] * p0[cidx[1]]
                     + cw[2] * p0[cidx[2]] + cw[3] * p0[cidx[3]];
            float v1 = cw[0] * p1[cidx[0]] + cw[1] * p1[cidx[1]]
                     + cw[2] * p1[cidx[2]] + cw[3] * p1[cidx[3]];
            PackU u; u.h[0] = (_Float16)v0; u.h[1] = (_Float16)v1;
            pk[i] = u.u;
        }
        // 4 x ds_store_b128: row gn, dwords [ghalf*16 .. ghalf*16+15]
        u32x4* wp = (u32x4*)(bufp + gn * LDSROW + ghalf * 16);
        #pragma unroll
        for (int j = 0; j < 4; ++j)
            wp[j] = (u32x4){pk[4*j], pk[4*j+1], pk[4*j+2], pk[4*j+3]};
    };

    gather(0, &ldsB[0][0]);
    __syncthreads();

    #pragma unroll 1
    for (int tap = 0; tap < K2; ++tap) {
        // Issue next tap's gather loads before this tap's WMMAs (latency overlap).
        if (tap + 1 < K2) gather(tap + 1, &ldsB[(tap + 1) & 1][0]);

        if (waveActive) {
            const uint32_t* bufp = &ldsB[tap & 1][0];
            #pragma unroll
            for (int s = 0; s < 2; ++s) {
                // A fragment (16x32 f16): lane m-row = ln15, contiguous k = 16*hi + e (matches B).
                FragU a;
                #pragma unroll
                for (int v = 0; v < 8; ++v) {
                    const int cpair = s * 16 + hi * 8 + v;
                    a.u[v] = wpk[((size_t)tap * 32 + cpair) * 64 + mtile * 16 + ln15];
                }
                #pragma unroll
                for (int nt = 0; nt < 4; ++nt) {
                    // B fragment (32x16 f16): lane n-col, 8 contiguous dwords
                    // -> 2 x ds_load_b128 per fragment.
                    const u32x4* bp = (const u32x4*)(bufp + (nt * 16 + ln15) * LDSROW
                                                     + s * 16 + hi * 8);
                    FragU bf;
                    bf.q[0] = bp[0];
                    bf.q[1] = bp[1];
                    acc[nt] = __builtin_amdgcn_wmma_f32_16x16x32_f16(
                        false, a.h, false, bf.h, (short)0, acc[nt], false, false);
                }
            }
        }
        __syncthreads();
    }

    // ---- epilogue: C/D layout -> lane = N col, vgpr v = M row (+8 if hi) ----
    if (!waveActive) return;
    #pragma unroll
    for (int nt = 0; nt < 4; ++nt) {
        const int n   = nBase + nt * 16 + ln15;
        const int rem = n - b * HW;
        #pragma unroll
        for (int v = 0; v < 8; ++v) {
            const int m = mtile * 16 + v + hi * 8;
            if (m >= Mout) continue;
            float val = acc[nt][v];
            const size_t oidx = ((size_t)b * Mout + m) * HW + rem;
            if (mode == 0) {
                out[oidx] = val + bias[m];
            } else if (mode == 1) {
                out[oidx] = relu_f(val * scale[m] + shift[m]);
            } else if (mode == 2) {
                out[oidx] = out[oidx] + relu_f(val * scale[m] + shift[m]);
            } else { // mode 3: bn + residual + relu
                const float r = residual[((size_t)b * CIN + m) * HW + rem];
                out[oidx] = relu_f(val * scale[m] + shift[m] + r);
            }
        }
    }
}

// ---------------------------------------------------------------------------
extern "C" void kernel_launch(void* const* d_in, const int* in_sizes, int n_in,
                              void* d_out, int out_size, void* d_ws, size_t ws_size,
                              hipStream_t stream) {
    const float* x   = (const float*)d_in[0];
    const float* ow1 = (const float*)d_in[1];
    const float* ob1 = (const float*)d_in[2];
    const float* ow3 = (const float*)d_in[3];
    const float* ob3 = (const float*)d_in[4];
    const float* ow2 = (const float*)d_in[5];
    const float* ob2 = (const float*)d_in[6];
    const float* w1  = (const float*)d_in[7];
    const float* w3  = (const float*)d_in[8];
    const float* w2  = (const float*)d_in[9];
    const float* g1 = (const float*)d_in[10], *b1 = (const float*)d_in[11];
    const float* m1 = (const float*)d_in[12], *v1 = (const float*)d_in[13];
    const float* g3 = (const float*)d_in[14], *b3 = (const float*)d_in[15];
    const float* m3 = (const float*)d_in[16], *v3 = (const float*)d_in[17];
    const float* g2 = (const float*)d_in[18], *b2 = (const float*)d_in[19];
    const float* m2 = (const float*)d_in[20], *v2 = (const float*)d_in[21];

    // Workspace carve-up (u32 units). Total ~15.6 MB.
    uint32_t* ws = (uint32_t*)d_ws;
    uint32_t* ow1pk = ws;                 // 9*32*64  = 18432
    uint32_t* ow3pk = ow1pk + 18432;
    uint32_t* ow2pk = ow3pk + 18432;
    uint32_t* w1pk  = ow2pk + 18432;
    uint32_t* w2pk  = w1pk  + 18432;
    uint32_t* w3pk  = w2pk  + 18432;      // 25*32*64 = 51200
    float*    bnp   = (float*)(w3pk + 51200);
    float* sc1 = bnp,        *sh1 = bnp + 64;
    float* sc3 = bnp + 128,  *sh3 = bnp + 192;
    float* sc2 = bnp + 256,  *sh2 = bnp + 320;
    float* off1 = bnp + 384;              // 8*18*3136 = 451584
    float* off3 = off1 + 451584;          // 8*50*3136 = 1254400
    float* off2 = off3 + 1254400;         // 451584
    float* mid  = off2 + 451584;          // 8*64*3136 = 1605632
    float* yout = (float*)d_out;

    // ---- prep: pack weights (f32 -> half2 A-fragment image), fold BN ----
    pack_w_kernel<<<144, 128, 0, stream>>>(ow1, ow1pk, 18, 9);
    pack_w_kernel<<<144, 128, 0, stream>>>(ow3, ow3pk, 50, 9);
    pack_w_kernel<<<144, 128, 0, stream>>>(ow2, ow2pk, 18, 9);
    pack_w_kernel<<<144, 128, 0, stream>>>(w1,  w1pk,  64, 9);
    pack_w_kernel<<<144, 128, 0, stream>>>(w2,  w2pk,  64, 9);
    pack_w_kernel<<<400, 128, 0, stream>>>(w3,  w3pk,  64, 25);
    bn_prep_kernel<<<1, 64, 0, stream>>>(g1, b1, m1, v1, sc1, sh1);
    bn_prep_kernel<<<1, 64, 0, stream>>>(g3, b3, m3, v3, sc3, sh3);
    bn_prep_kernel<<<1, 64, 0, stream>>>(g2, b2, m2, v2, sc2, sh2);

    const int NBLK = (8 * HW) / 64;   // 25088 / 64 = 392

    // 1) offset1 = conv3x3(x) + ob1
    deform_conv_wmma<3, 3, false><<<NBLK, 128, 0, stream>>>(
        x, nullptr, ow1pk, off1, 18, 1, 1, 0, ob1, nullptr, nullptr, nullptr);
    // 2) offset3 = conv3x3(x) + ob3
    deform_conv_wmma<3, 3, false><<<NBLK, 128, 0, stream>>>(
        x, nullptr, ow3pk, off3, 50, 1, 1, 0, ob3, nullptr, nullptr, nullptr);
    // 3) mid = relu(bn1(deform3x3(x, offset1), pad=2, dil=2))
    deform_conv_wmma<3, 3, true><<<NBLK, 128, 0, stream>>>(
        x, off1, w1pk, mid, 64, 2, 2, 1, nullptr, sc1, sh1, nullptr);
    // 4) mid += relu(bn3(deform5x5(x, offset3), pad=4, dil=2))
    deform_conv_wmma<5, 5, true><<<NBLK, 128, 0, stream>>>(
        x, off3, w3pk, mid, 64, 4, 2, 2, nullptr, sc3, sh3, nullptr);
    // 5) offset2 = conv3x3(mid) + ob2
    deform_conv_wmma<3, 3, false><<<NBLK, 128, 0, stream>>>(
        mid, nullptr, ow2pk, off2, 18, 1, 1, 0, ob2, nullptr, nullptr, nullptr);
    // 6) out = relu(bn2(deform3x3(mid, offset2), pad=2, dil=2) + x)
    deform_conv_wmma<3, 3, true><<<NBLK, 128, 0, stream>>>(
        mid, off2, w2pk, yout, 64, 2, 2, 3, nullptr, sc2, sh2, x);
}
